// BERT_17549236372164
// MI455X (gfx1250) — compile-verified, matrix operand order checked by reading, not run on hardware
//
#include <hip/hip_runtime.h>

// ---------------- problem constants (BERT-base, B=16, S=512) ----------------
constexpr int   Hh  = 768;
constexpr int   Ll  = 12;
constexpr int   NHh = 12;
constexpr int   Ii  = 3072;
constexpr int   Ss  = 512;
constexpr int   Bb  = 16;
constexpr int   DHh = 64;
constexpr int   Mm  = Bb * Ss;     // 8192 rows
constexpr float EPS = 1e-12f;

typedef __attribute__((ext_vector_type(16))) __bf16        v16bf;
typedef __attribute__((ext_vector_type(8)))  float         v8f;
typedef __attribute__((ext_vector_type(4)))  unsigned int  v4u;
typedef __attribute__((ext_vector_type(8)))  unsigned int  v8u;

// ---------------- helpers ----------------
static __device__ inline unsigned short f2bf(float f) {
  union { float f; unsigned int u; } x; x.f = f;
  unsigned int u = x.u;
  unsigned int r = (u + 0x7FFFu + ((u >> 16) & 1u)) >> 16;   // RNE
  return (unsigned short)r;
}

static __device__ inline float gelu_exact(float x) {
  return 0.5f * x * (1.0f + erff(x * 0.70710678118654752f));
}

// Async copy 16B global -> LDS (GLOBAL_LOAD_ASYNC_TO_LDS_B128, ASYNCcnt).
// ldsp: generic pointer into __shared__; low 32 bits of the flat address are
// the LDS byte offset (LDS aperture maps addr[31:0] -> LDS_ADDR).
static __device__ inline void async_cp16(void* ldsp, const void* gp) {
  unsigned int loff = (unsigned int)(unsigned long long)ldsp;
  asm volatile("global_load_async_to_lds_b128 %0, %1, off"
               :: "v"(loff), "v"(gp)
               : "memory");
}
static __device__ inline void async_wait0() {
  asm volatile("s_wait_asynccnt 0x0" ::: "memory");
}

// Tensor Data Mover: DMA a 2D tile of 2-byte elements (rows x cols, global row
// stride rowStrideElems) into LDS at ldsp. Issue from ONE wave only; pair with
// s_wait_tensorcnt(0) from that wave before the workgroup barrier.
// D# group 0: count=1 | lds_addr | global_addr | type=2 ("image").
// D# group 1: data_size=2B, tensor_dim0/1 (OOB-safe), tile_dim0=cols,
//             tile_dim1=rows, tensor_dim0_stride=rowStrideElems.
static __device__ inline void tdm_load_tile_2d(void* ldsp, const void* gp,
                                               int rows, int cols,
                                               int rowStrideElems) {
  unsigned long long ga = (unsigned long long)gp;
  v4u g0;
  g0[0] = 1u;                                        // count=1 (valid D#)
  g0[1] = (unsigned int)(unsigned long long)ldsp;    // lds byte address
  g0[2] = (unsigned int)ga;                          // global addr [31:0]
  g0[3] = (unsigned int)(ga >> 32) | (2u << 30);     // addr[56:32] | type=2
  unsigned int td0 = (unsigned int)rowStrideElems;   // tensor dim0 (elems)
  unsigned int td1 = (unsigned int)rows;             // tensor dim1 (rows)
  v8u g1;
  g1[0] = 1u << 16;                                  // data_size=1 (2 bytes)
  g1[1] = (td0 & 0xFFFFu) << 16;                     // tensor_dim0 [15:0]
  g1[2] = (td0 >> 16) | ((td1 & 0xFFFFu) << 16);     // dim0 hi | dim1 lo
  g1[3] = (td1 >> 16) | (((unsigned int)cols) << 16);// dim1 hi | tile_dim0
  g1[4] = (unsigned int)rows & 0xFFFFu;              // tile_dim1, tile_dim2=0
  g1[5] = (unsigned int)rowStrideElems;              // tensor_dim0_stride lo32
  g1[6] = 0u;                                        // stride hi | dim1_stride
  g1[7] = 0u;
  asm volatile("tensor_load_to_lds %0, %1" :: "s"(g0), "s"(g1) : "memory");
}
static __device__ inline void tensor_wait0() {
  __builtin_amdgcn_s_wait_tensorcnt(0);
}

// Load a 16x32 bf16 A/B fragment from LDS laid out row-major [row][ldk].
// Lane l holds row (base + l%16); hi = l/16 selects K sub-runs:
//   elements 0..7  -> K = koff + 8*hi + 0..7
//   elements 8..15 -> K = koff + 16 + 8*hi + 0..7
static __device__ inline v16bf load_frag(const unsigned short* lds, int row,
                                         int ldk, int koff, int hi) {
  union { v16bf v; uint4 q[2]; } u;
  const unsigned short* p = lds + row * ldk + koff + 8 * hi;
  u.q[0] = *(const uint4*)(p);
  u.q[1] = *(const uint4*)(p + 16);
  return u.v;
}

static __device__ inline v8f wmma_bf16(v16bf a, v16bf b, v8f c) {
  return __builtin_amdgcn_wmma_f32_16x16x32_bf16(
      false, a, false, b, (short)0, c, false, false);
}

// ---------------- f32 -> bf16 conversion ----------------
__global__ void f32_to_bf16_kernel(const float* __restrict__ in,
                                   unsigned short* __restrict__ out, size_t n) {
  size_t i = (size_t)blockIdx.x * blockDim.x + threadIdx.x;
  size_t stride = (size_t)gridDim.x * blockDim.x;
  for (; i < n; i += stride) out[i] = f2bf(in[i]);
}

// ---------------- embedding gather + layernorm ----------------
__global__ __launch_bounds__(256) void embed_ln_kernel(
    const int* __restrict__ ids, const int* __restrict__ tts,
    const float* __restrict__ word, const float* __restrict__ pos,
    const float* __restrict__ type, const float* __restrict__ lnw,
    const float* __restrict__ lnb, float* __restrict__ outF,
    unsigned short* __restrict__ outB) {
  int m = blockIdx.x;               // token row
  int tid = threadIdx.x;
  int s = m & (Ss - 1);
  int id = ids[m];
  int tt = tts[m];
  __shared__ float red[256];

  float x[3];
  #pragma unroll
  for (int j = 0; j < 3; j++) {
    int c = tid + j * 256;
    x[j] = word[(size_t)id * Hh + c] + pos[(size_t)s * Hh + c] +
           type[(size_t)tt * Hh + c];
  }
  float ls = x[0] + x[1] + x[2];
  red[tid] = ls; __syncthreads();
  for (int o = 128; o > 0; o >>= 1) { if (tid < o) red[tid] += red[tid + o]; __syncthreads(); }
  float mean = red[0] * (1.0f / Hh);
  __syncthreads();
  float lv = 0.f;
  #pragma unroll
  for (int j = 0; j < 3; j++) { float d = x[j] - mean; lv += d * d; }
  red[tid] = lv; __syncthreads();
  for (int o = 128; o > 0; o >>= 1) { if (tid < o) red[tid] += red[tid + o]; __syncthreads(); }
  float rstd = rsqrtf(red[0] * (1.0f / Hh) + EPS);
  #pragma unroll
  for (int j = 0; j < 3; j++) {
    int c = tid + j * 256;
    float v = lnw[c] * (x[j] - mean) * rstd + lnb[c];
    outF[(size_t)m * Hh + c] = v;
    outB[(size_t)m * Hh + c] = f2bf(v);
  }
}

// ---------------- layernorm (post-attn / post-ffn) ----------------
__global__ __launch_bounds__(256) void ln_kernel(
    const float* __restrict__ xin, const float* __restrict__ lnw,
    const float* __restrict__ lnb, float* __restrict__ outF,
    unsigned short* __restrict__ outB) {
  int m = blockIdx.x;
  int tid = threadIdx.x;
  __shared__ float red[256];
  float x[3];
  #pragma unroll
  for (int j = 0; j < 3; j++) x[j] = xin[(size_t)m * Hh + tid + j * 256];
  float ls = x[0] + x[1] + x[2];
  red[tid] = ls; __syncthreads();
  for (int o = 128; o > 0; o >>= 1) { if (tid < o) red[tid] += red[tid + o]; __syncthreads(); }
  float mean = red[0] * (1.0f / Hh);
  __syncthreads();
  float lv = 0.f;
  #pragma unroll
  for (int j = 0; j < 3; j++) { float d = x[j] - mean; lv += d * d; }
  red[tid] = lv; __syncthreads();
  for (int o = 128; o > 0; o >>= 1) { if (tid < o) red[tid] += red[tid + o]; __syncthreads(); }
  float rstd = rsqrtf(red[0] * (1.0f / Hh) + EPS);
  #pragma unroll
  for (int j = 0; j < 3; j++) {
    int c = tid + j * 256;
    float v = lnw[c] * (x[j] - mean) * rstd + lnb[c];
    outF[(size_t)m * Hh + c] = v;
    outB[(size_t)m * Hh + c] = f2bf(v);
  }
}

// ---------------- tiled bf16 WMMA GEMM: C = A[M,K] x W[K,N] + bias ----------
// Double-buffered LDS; A tile staged by the Tensor Data Mover (one
// tensor_load_to_lds per stage, issued by wave 0, TENSORcnt-tracked); B tile
// transposed through VGPRs (packed dword stores).
// MODE 0: f32 out = acc + bias + resid          (attn-out, FFN2 pre-LN)
// MODE 1: bf16 out = gelu(acc + bias)           (FFN1)
// MODE 2: bf16 out scattered to [B,NH,S,DH]     (Q/K/V projections, N==768)
template <int MODE>
__global__ __launch_bounds__(256) void gemm_bf16_kernel(
    const unsigned short* __restrict__ A, const unsigned short* __restrict__ W,
    const float* __restrict__ bias, const float* __restrict__ resid,
    float* __restrict__ outF, unsigned short* __restrict__ outB,
    int Mdim, int Ndim, int Kdim) {
  constexpr int BM = 128, BN = 128, BK = 32;
  __shared__ unsigned short Alds[2][BM * BK];   // [m][k]
  __shared__ unsigned short Bt[2][BN * BK];     // [n][k]  (W transposed)

  int tid = threadIdx.x;
  int lane = tid & 31, wv = tid >> 5;
  int wm = wv & 1, wn = wv >> 1;             // 2 waves on M, 4 on N
  int lo = lane & 15, hi = lane >> 4;
  int mbase = blockIdx.y * BM, nbase = blockIdx.x * BN;

  v8f acc[4][2];
  #pragma unroll
  for (int mi = 0; mi < 4; mi++)
    #pragma unroll
    for (int ni = 0; ni < 2; ni++) acc[mi][ni] = (v8f){0,0,0,0,0,0,0,0};

  int krp = tid >> 4, seg = tid & 15;        // B: k-pair 2*krp, 8 n-cols each

  auto stageA = [&](int buf, int k0) {       // TDM: whole 128x32 tile, 1 inst
    if (wv == 0)
      tdm_load_tile_2d(&Alds[buf][0],
                       A + (size_t)mbase * Kdim + k0, BM, BK, Kdim);
  };
  auto stageB = [&](int buf, int k0) {
    union { uint4 q; unsigned short u[8]; } r0, r1;
    r0.q = *(const uint4*)(W + (size_t)(k0 + 2 * krp) * Ndim + nbase + seg * 8);
    r1.q = *(const uint4*)(W + (size_t)(k0 + 2 * krp + 1) * Ndim + nbase + seg * 8);
    unsigned int* bt32 = (unsigned int*)&Bt[buf][0];
    #pragma unroll
    for (int e = 0; e < 8; e++)
      bt32[((seg * 8 + e) * BK + 2 * krp) >> 1] =
          (unsigned int)r0.u[e] | ((unsigned int)r1.u[e] << 16);
  };

  int nk = Kdim / BK;
  int cur = 0;
  stageA(0, 0);
  stageB(0, 0);
  if (wv == 0) tensor_wait0();
  __syncthreads();

  for (int kt = 0; kt < nk; kt++) {
    int nxt = cur ^ 1;
    if (kt + 1 < nk) { stageA(nxt, (kt + 1) * BK); stageB(nxt, (kt + 1) * BK); }

    v16bf af[4], bfr[2];
    #pragma unroll
    for (int mi = 0; mi < 4; mi++)
      af[mi] = load_frag(&Alds[cur][0], wm * 64 + mi * 16 + lo, BK, 0, hi);
    #pragma unroll
    for (int ni = 0; ni < 2; ni++)
      bfr[ni] = load_frag(&Bt[cur][0], wn * 32 + ni * 16 + lo, BK, 0, hi);
    #pragma unroll
    for (int mi = 0; mi < 4; mi++)
      #pragma unroll
      for (int ni = 0; ni < 2; ni++)
        acc[mi][ni] = wmma_bf16(af[mi], bfr[ni], acc[mi][ni]);

    if (wv == 0) tensor_wait0();
    __syncthreads();
    cur = nxt;
  }

  // epilogue (C/D layout: lane -> col, vgpr r + 8*hi -> row)
  #pragma unroll
  for (int mi = 0; mi < 4; mi++)
    #pragma unroll
    for (int ni = 0; ni < 2; ni++)
      #pragma unroll
      for (int r = 0; r < 8; r++) {
        int row = mbase + wm * 64 + mi * 16 + r + 8 * hi;
        int col = nbase + wn * 32 + ni * 16 + lo;
        float v = acc[mi][ni][r] + bias[col];
        if (MODE == 0) {
          v += resid[(size_t)row * Ndim + col];
          outF[(size_t)row * Ndim + col] = v;
        } else if (MODE == 1) {
          outB[(size_t)row * Ndim + col] = f2bf(gelu_exact(v));
        } else {  // QKV scatter: [B,NH,S,DH]
          int b = row >> 9, s = row & 511;
          int hh = col >> 6, dh = col & 63;
          outB[(((size_t)(b * NHh + hh) * Ss) + s) * DHh + dh] = f2bf(v);
        }
      }
}

// ---------------- flash attention (per 64-query tile, per head) -------------
__global__ __launch_bounds__(128) void attn_kernel(
    const unsigned short* __restrict__ Qb, const unsigned short* __restrict__ Kb,
    const unsigned short* __restrict__ Vb, const float* __restrict__ amask,
    unsigned short* __restrict__ ctxB) {
  __shared__ unsigned short Qs[64 * 64];
  __shared__ unsigned short Ks[64 * 64];
  __shared__ unsigned short Vt[64 * 64];   // [dh][key]
  __shared__ unsigned short Ps[64 * 64];   // per-wave 16-row regions

  int tid = threadIdx.x, lane = tid & 31, w = tid >> 5;
  int lo = lane & 15, hi = lane >> 4;
  int qt = blockIdx.x, h = blockIdx.y, b = blockIdx.z;
  size_t head_base = ((size_t)(b * NHh + h)) * Ss * DHh;
  int q0 = qt * 64;

  int row = tid >> 1, half = tid & 1;      // Q/K tiles: 2 threads/row, 32 elems
  int kp = tid >> 2, qd = (tid & 3) * 16;  // V tile: key pair 2*kp, 16 dh each

  // load Q tile [64 q][64 dh] (async to LDS)
  {
    const unsigned short* gq =
        Qb + head_base + (size_t)(q0 + row) * DHh + half * 32;
    unsigned short* lq = &Qs[row * 64 + half * 32];
    #pragma unroll
    for (int j = 0; j < 4; j++) async_cp16(lq + j * 8, gq + j * 8);
  }

  float mrow[8], lsum[8];
  #pragma unroll
  for (int r = 0; r < 8; r++) { mrow[r] = -1e30f; lsum[r] = 0.f; }
  v8f cacc[4];
  #pragma unroll
  for (int ni = 0; ni < 4; ni++) cacc[ni] = (v8f){0,0,0,0,0,0,0,0};

  for (int kb = 0; kb < Ss / 64; kb++) {
    int key0 = kb * 64;
    __syncthreads();   // previous iteration done with Ks/Vt/Ps
    // load K tile [64 key][64 dh] (async to LDS)
    {
      const unsigned short* gk =
          Kb + head_base + (size_t)(key0 + row) * DHh + half * 32;
      unsigned short* lk = &Ks[row * 64 + half * 32];
      #pragma unroll
      for (int j = 0; j < 4; j++) async_cp16(lk + j * 8, gk + j * 8);
    }
    // load V tile transposed -> Vt[dh][key], two keys packed per dword
    {
      union { uint4 q[2]; unsigned short u[16]; } r0, r1;
      const uint4* g0 =
          (const uint4*)(Vb + head_base + (size_t)(key0 + 2 * kp) * DHh + qd);
      const uint4* g1 =
          (const uint4*)(Vb + head_base + (size_t)(key0 + 2 * kp + 1) * DHh + qd);
      r0.q[0] = g0[0]; r0.q[1] = g0[1];
      r1.q[0] = g1[0]; r1.q[1] = g1[1];
      unsigned int* vt32 = (unsigned int*)Vt;
      #pragma unroll
      for (int e = 0; e < 16; e++)
        vt32[((qd + e) * 64 + 2 * kp) >> 1] =
            (unsigned int)r0.u[e] | ((unsigned int)r1.u[e] << 16);
    }
    async_wait0();
    __syncthreads();

    // scores: S = Q(16 q rows of this wave) x K^T, contraction over dh=64
    v8f sacc[4];
    #pragma unroll
    for (int nt = 0; nt < 4; nt++) sacc[nt] = (v8f){0,0,0,0,0,0,0,0};
    #pragma unroll
    for (int ks = 0; ks < 2; ks++) {
      v16bf aq = load_frag(Qs, w * 16 + lo, 64, ks * 32, hi);
      #pragma unroll
      for (int nt = 0; nt < 4; nt++) {
        v16bf bk = load_frag(Ks, nt * 16 + lo, 64, ks * 32, hi);
        sacc[nt] = wmma_bf16(aq, bk, sacc[nt]);
      }
    }
    float mk[4];
    #pragma unroll
    for (int nt = 0; nt < 4; nt++) {
      float am = amask[(size_t)b * Ss + key0 + nt * 16 + lo];
      mk[nt] = (1.0f - am) * -10000.0f;
    }
    // online softmax per query row (row = r + 8*hi of this wave)
    #pragma unroll
    for (int r = 0; r < 8; r++) {
      float sv[4], mx = -1e30f;
      #pragma unroll
      for (int nt = 0; nt < 4; nt++) {
        sv[nt] = sacc[nt][r] * 0.125f + mk[nt];
        mx = fmaxf(mx, sv[nt]);
      }
      #pragma unroll
      for (int o = 1; o < 16; o <<= 1) mx = fmaxf(mx, __shfl_xor(mx, o, 32));
      float mnew = fmaxf(mrow[r], mx);
      float scale = __expf(mrow[r] - mnew);
      float ssum = 0.f;
      #pragma unroll
      for (int nt = 0; nt < 4; nt++) {
        float p = __expf(sv[nt] - mnew);
        ssum += p;
        Ps[(w * 16 + r + 8 * hi) * 64 + nt * 16 + lo] = f2bf(p);
      }
      #pragma unroll
      for (int o = 1; o < 16; o <<= 1) ssum += __shfl_xor(ssum, o, 32);
      lsum[r] = lsum[r] * scale + ssum;
      mrow[r] = mnew;
      #pragma unroll
      for (int ni = 0; ni < 4; ni++) cacc[ni][r] *= scale;
    }
    __syncthreads();   // Ps visible

    // ctx += P x V  (contraction over 64 keys)
    #pragma unroll
    for (int ks = 0; ks < 2; ks++) {
      v16bf ap = load_frag(Ps, w * 16 + lo, 64, ks * 32, hi);
      #pragma unroll
      for (int ni = 0; ni < 4; ni++) {
        v16bf bv = load_frag(Vt, ni * 16 + lo, 64, ks * 32, hi);
        cacc[ni] = wmma_bf16(ap, bv, cacc[ni]);
      }
    }
  }

  // write ctx back as [M, H] bf16 (col = h*64 + dh)
  #pragma unroll
  for (int ni = 0; ni < 4; ni++)
    #pragma unroll
    for (int r = 0; r < 8; r++) {
      int q = q0 + w * 16 + r + 8 * hi;
      int m = b * Ss + q;
      int col = h * DHh + ni * 16 + lo;
      ctxB[(size_t)m * Hh + col] = f2bf(cacc[ni][r] / lsum[r]);
    }
}

// ---------------- host orchestration ----------------
extern "C" void kernel_launch(void* const* d_in, const int* in_sizes, int n_in,
                              void* d_out, int out_size, void* d_ws,
                              size_t ws_size, hipStream_t stream) {
  const int*   input_ids = (const int*)d_in[0];
  const int*   tok_type  = (const int*)d_in[1];
  const float* amask     = (const float*)d_in[2];
  const float* word      = (const float*)d_in[3];
  const float* pos       = (const float*)d_in[4];
  const float* type      = (const float*)d_in[5];
  const float* elnw      = (const float*)d_in[6];
  const float* elnb      = (const float*)d_in[7];
  const float* Wq = (const float*)d_in[8];  const float* bq = (const float*)d_in[9];
  const float* Wk = (const float*)d_in[10]; const float* bk = (const float*)d_in[11];
  const float* Wv = (const float*)d_in[12]; const float* bv = (const float*)d_in[13];
  const float* Wo = (const float*)d_in[14]; const float* bo = (const float*)d_in[15];
  const float* ln1w = (const float*)d_in[16]; const float* ln1b = (const float*)d_in[17];
  const float* W1 = (const float*)d_in[18]; const float* b1 = (const float*)d_in[19];
  const float* W2 = (const float*)d_in[20]; const float* b2 = (const float*)d_in[21];
  const float* ln2w = (const float*)d_in[22]; const float* ln2b = (const float*)d_in[23];

  const size_t szHH = (size_t)Ll * Hh * Hh;   // per QKVO weight stack
  const size_t szHI = (size_t)Ll * Hh * Ii;   // W1 / W2 stacks
  const size_t MH   = (size_t)Mm * Hh;
  const size_t MI   = (size_t)Mm * Ii;

  char* ws = (char*)d_ws;
  size_t off = 0;
  auto alloc = [&](size_t bytes) -> void* {
    void* p = ws + off;
    off = (off + bytes + 255) & ~(size_t)255;
    return p;
  };
  unsigned short* wqB = (unsigned short*)alloc(szHH * 2);
  unsigned short* wkB = (unsigned short*)alloc(szHH * 2);
  unsigned short* wvB = (unsigned short*)alloc(szHH * 2);
  unsigned short* woB = (unsigned short*)alloc(szHH * 2);
  unsigned short* w1B = (unsigned short*)alloc(szHI * 2);
  unsigned short* w2B = (unsigned short*)alloc(szHI * 2);
  float*          hF   = (float*)alloc(MH * 4);
  float*          tmpF = (float*)alloc(MH * 4);
  unsigned short* xB   = (unsigned short*)alloc(MH * 2);
  unsigned short* qB   = (unsigned short*)alloc(MH * 2);
  unsigned short* kB   = (unsigned short*)alloc(MH * 2);
  unsigned short* vB   = (unsigned short*)alloc(MH * 2);
  unsigned short* cB   = (unsigned short*)alloc(MH * 2);
  unsigned short* gB   = (unsigned short*)alloc(MI * 2);

  // weights fp32 -> bf16
  f32_to_bf16_kernel<<<2048, 256, 0, stream>>>(Wq, wqB, szHH);
  f32_to_bf16_kernel<<<2048, 256, 0, stream>>>(Wk, wkB, szHH);
  f32_to_bf16_kernel<<<2048, 256, 0, stream>>>(Wv, wvB, szHH);
  f32_to_bf16_kernel<<<2048, 256, 0, stream>>>(Wo, woB, szHH);
  f32_to_bf16_kernel<<<2048, 256, 0, stream>>>(W1, w1B, szHI);
  f32_to_bf16_kernel<<<2048, 256, 0, stream>>>(W2, w2B, szHI);

  // embeddings + LN
  embed_ln_kernel<<<Mm, 256, 0, stream>>>(input_ids, tok_type, word, pos, type,
                                          elnw, elnb, hF, xB);

  dim3 gH(Hh / 128, Mm / 128);   // (6, 64)
  dim3 gI(Ii / 128, Mm / 128);   // (24, 64)
  dim3 gA(Ss / 64, NHh, Bb);     // (8, 12, 16)

  for (int l = 0; l < Ll; l++) {
    const unsigned short* wq_l = wqB + (size_t)l * Hh * Hh;
    const unsigned short* wk_l = wkB + (size_t)l * Hh * Hh;
    const unsigned short* wv_l = wvB + (size_t)l * Hh * Hh;
    const unsigned short* wo_l = woB + (size_t)l * Hh * Hh;
    const unsigned short* w1_l = w1B + (size_t)l * Hh * Ii;
    const unsigned short* w2_l = w2B + (size_t)l * Ii * Hh;
    const float* bq_l = bq + (size_t)l * Hh;
    const float* bk_l = bk + (size_t)l * Hh;
    const float* bv_l = bv + (size_t)l * Hh;
    const float* bo_l = bo + (size_t)l * Hh;
    const float* b1_l = b1 + (size_t)l * Ii;
    const float* b2_l = b2 + (size_t)l * Hh;
    const float* l1w = ln1w + (size_t)l * Hh; const float* l1b = ln1b + (size_t)l * Hh;
    const float* l2w = ln2w + (size_t)l * Hh; const float* l2b = ln2b + (size_t)l * Hh;

    // Q/K/V projections (bf16 out, head-scattered)
    gemm_bf16_kernel<2><<<gH, 256, 0, stream>>>(xB, wq_l, bq_l, nullptr,
                                                nullptr, qB, Mm, Hh, Hh);
    gemm_bf16_kernel<2><<<gH, 256, 0, stream>>>(xB, wk_l, bk_l, nullptr,
                                                nullptr, kB, Mm, Hh, Hh);
    gemm_bf16_kernel<2><<<gH, 256, 0, stream>>>(xB, wv_l, bv_l, nullptr,
                                                nullptr, vB, Mm, Hh, Hh);
    // attention
    attn_kernel<<<gA, 128, 0, stream>>>(qB, kB, vB, amask, cB);
    // attn-out projection + residual, then LN1 -> (hF, xB)
    gemm_bf16_kernel<0><<<gH, 256, 0, stream>>>(cB, wo_l, bo_l, hF, tmpF,
                                                nullptr, Mm, Hh, Hh);
    ln_kernel<<<Mm, 256, 0, stream>>>(tmpF, l1w, l1b, hF, xB);
    // FFN1 (GELU, bf16 out)
    gemm_bf16_kernel<1><<<gI, 256, 0, stream>>>(xB, w1_l, b1_l, nullptr,
                                                nullptr, gB, Mm, Ii, Hh);
    // FFN2 + residual, then LN2 -> (hF or d_out, xB)
    gemm_bf16_kernel<0><<<gH, 256, 0, stream>>>(gB, w2_l, b2_l, hF, tmpF,
                                                nullptr, Mm, Hh, Ii);
    float* lnout = (l == Ll - 1) ? (float*)d_out : hF;
    ln_kernel<<<Mm, 256, 0, stream>>>(tmpF, l2w, l2b, lnout, xB);
  }
}
